// OmostDenseDiffusionCrossAttention_67156108640389
// MI455X (gfx1250) — matrix operand order repaired and checked by reading, not compiled, not run on hardware
//
#include <hip/hip_runtime.h>

// -------- CDNA5 (gfx1250) wave32 WMMA flash-attention for DenseDiffusion masked SDPA --------
// q:[B,SQ,D] f32, k/v:[B,SKV,D] f32, mask_bool:[B,H,SQ,SKV] u8, mask_scale:[B,H,SQ,SKV] f32
// out:[B,SQ,D] f32.  D = H*DH, H=10, DH=64.
//
// Memory plan (MI455X, 23.3 TB/s HBM, 192 MB L2):
//  - mask_scale/mask_bool: ~189 MB streamed once -> NT loads (don't churn L2)
//  - K/V: 4.8 MB, reused by 32 q-tile workgroups per (b,h) -> regular temporal loads
//  - q / out: read/written once -> NT load / NT store

typedef __attribute__((ext_vector_type(16))) __bf16        v16bf;
typedef __attribute__((ext_vector_type(8)))  float         v8f;
typedef __attribute__((ext_vector_type(4)))  unsigned int  u32x4;

union Frag16 {
    v16bf          v;
    unsigned short u[16];
    u32x4          q[2];
};

static __device__ __forceinline__ unsigned short f2bf(float f) {
    union { float f; unsigned int u; } x; x.f = f;
    unsigned int r = 0x7FFFu + ((x.u >> 16) & 1u);   // round to nearest even
    x.u += r;
    return (unsigned short)(x.u >> 16);
}

#define LOG2E 1.44269504088896340736f

constexpr int Bb      = 2;
constexpr int SQ      = 4096;
constexpr int SKV     = 462;
constexpr int H       = 10;
constexpr int DH      = 64;
constexpr int Dd      = H * DH;      // 640
constexpr int SKV_PAD = 480;         // multiple of 32
constexpr int QTILE   = 128;         // query rows per workgroup
constexpr int NW      = 8;           // waves per workgroup (256 threads)
constexpr int NTHR    = NW * 32;

// LDS layout (ushort units)
constexpr int K_OFF  = 0;                        // K  : [SKV_PAD][DH]   bf16
constexpr int VT_OFF = K_OFF + SKV_PAD * DH;     // Vt : [DH][SKV_PAD]   bf16 (transposed)
constexpr int P_OFF  = VT_OFF + DH * SKV_PAD;    // P  : [NW][16][32]    bf16
constexpr int LDS_US = P_OFF + NW * 16 * 32;     // 65536 ushorts = 128 KB

__global__ void __launch_bounds__(NTHR)
omost_dd_attn_kernel(const float* __restrict__ qg,
                     const float* __restrict__ kg,
                     const float* __restrict__ vg,
                     const unsigned char* __restrict__ mask_bool,
                     const float* __restrict__ mask_scale,
                     float* __restrict__ out)
{
    extern __shared__ __align__(16) unsigned short smem[];
    unsigned short* K_lds  = smem + K_OFF;
    unsigned short* Vt_lds = smem + VT_OFF;
    unsigned short* P_lds  = smem + P_OFF;

    const int tid  = threadIdx.x;
    const int w    = tid >> 5;         // wave id 0..7
    const int lane = tid & 31;
    const int ln16 = lane & 15;        // col within 16-group / A-row
    const int kh   = lane >> 4;        // half-select (A K-half, B K-half, C row-half)

    const int b  = blockIdx.z;
    const int h  = blockIdx.y;
    const int q0 = blockIdx.x * QTILE;

    // ---------------- cooperative K/V -> LDS (bf16, V transposed, zero padding) ----------------
    const float* kgb = kg + (size_t)b * SKV * Dd + h * DH;
    const float* vgb = vg + (size_t)b * SKV * Dd + h * DH;
    for (int idx = tid; idx < SKV_PAD * DH; idx += NTHR) {
        const int kv = idx >> 6;         // / DH
        const int dh = idx & 63;         // % DH
        float kvv = 0.0f, vvv = 0.0f;
        if (kv < SKV) {
            kvv = kgb[(size_t)kv * Dd + dh];   // temporal: reused by other workgroups via L2
            vvv = vgb[(size_t)kv * Dd + dh];
        }
        K_lds[kv * DH + dh]        = f2bf(kvv);
        Vt_lds[dh * SKV_PAD + kv]  = f2bf(vvv);
    }
    __syncthreads();

    // ---------------- per-wave: load 16 query rows into two A fragments (pre-scaled) ----------------
    const int qrow_a = q0 + w * 16 + ln16;              // A-layout row for this lane
    const float* qrow = qg + ((size_t)b * SQ + qrow_a) * Dd + h * DH;
    Frag16 aq0, aq1;
#pragma unroll
    for (int i = 0; i < 8; ++i) {
        const int d0 = kh * 16 + i;
        aq0.u[i]     = f2bf(__builtin_nontemporal_load(qrow + d0)      * 0.125f); // K = 16*kh + 0..7
        aq0.u[i + 8] = f2bf(__builtin_nontemporal_load(qrow + 8 + d0)  * 0.125f); // K = 16*kh + 8..15
        aq1.u[i]     = f2bf(__builtin_nontemporal_load(qrow + 32 + d0) * 0.125f);
        aq1.u[i + 8] = f2bf(__builtin_nontemporal_load(qrow + 40 + d0) * 0.125f);
    }

    // online-softmax state: this lane tracks rows (r + 8*kh), r = 0..7
    float mrow[8], lrow[8];
#pragma unroll
    for (int r = 0; r < 8; ++r) { mrow[r] = -1e30f; lrow[r] = 0.0f; }
    v8f o0 = {}, o1 = {}, o2 = {}, o3 = {};

    // mask base for this wave's 16-row block
    const size_t mbase = (((size_t)b * H + h) * SQ + (q0 + w * 16)) * SKV;
    const unsigned char* mb = mask_bool  + mbase;
    const float*         ms = mask_scale + mbase;

    unsigned short* pw = P_lds + w * 512;               // per-wave P staging [16][32]

    // ---------------- main loop over keys, 32 at a time ----------------
    for (int n0 = 0; n0 < SKV_PAD; n0 += 32) {
        // ---- scores: two 16x16 tiles, K-dim = DH = 64 (2 WMMA each) ----
        v8f c0 = {}, c1 = {};
        {
            const unsigned short* kr = &K_lds[(n0 + ln16) * DH];
            Frag16 b0, b1;
            b0.q[0] = *(const u32x4*)(kr + kh * 16);
            b0.q[1] = *(const u32x4*)(kr + kh * 16 + 8);
            b1.q[0] = *(const u32x4*)(kr + 32 + kh * 16);
            b1.q[1] = *(const u32x4*)(kr + 32 + kh * 16 + 8);
            c0 = __builtin_amdgcn_wmma_f32_16x16x32_bf16(false, aq0.v, false, b0.v, (short)0, c0, false, false);
            c0 = __builtin_amdgcn_wmma_f32_16x16x32_bf16(false, aq1.v, false, b1.v, (short)0, c0, false, false);
        }
        {
            const unsigned short* kr = &K_lds[(n0 + 16 + ln16) * DH];
            Frag16 b0, b1;
            b0.q[0] = *(const u32x4*)(kr + kh * 16);
            b0.q[1] = *(const u32x4*)(kr + kh * 16 + 8);
            b1.q[0] = *(const u32x4*)(kr + 32 + kh * 16);
            b1.q[1] = *(const u32x4*)(kr + 32 + kh * 16 + 8);
            c1 = __builtin_amdgcn_wmma_f32_16x16x32_bf16(false, aq0.v, false, b0.v, (short)0, c1, false, false);
            c1 = __builtin_amdgcn_wmma_f32_16x16x32_bf16(false, aq1.v, false, b1.v, (short)0, c1, false, false);
        }

        // ---- mask + scale (C layout: row = r + 8*kh, col = ln16); NT streaming loads ----
        float p0[8], p1[8];
        const int col0 = n0 + ln16;
        const int col1 = n0 + 16 + ln16;
#pragma unroll
        for (int r = 0; r < 8; ++r) {
            const size_t moff = (size_t)(r + 8 * kh) * SKV;
            float s0 = -1e30f, s1 = -1e30f;
            if (col0 < SKV && __builtin_nontemporal_load(mb + moff + col0))
                s0 = c0[r] * __builtin_nontemporal_load(ms + moff + col0);
            if (col1 < SKV && __builtin_nontemporal_load(mb + moff + col1))
                s1 = c1[r] * __builtin_nontemporal_load(ms + moff + col1);
            p0[r] = s0; p1[r] = s1;
        }

        // ---- online softmax update (reduce across the 16-lane row group) ----
#pragma unroll
        for (int r = 0; r < 8; ++r) {
            float t = fmaxf(p0[r], p1[r]);
            t = fmaxf(t, __shfl_xor(t, 1, 32));
            t = fmaxf(t, __shfl_xor(t, 2, 32));
            t = fmaxf(t, __shfl_xor(t, 4, 32));
            t = fmaxf(t, __shfl_xor(t, 8, 32));
            const float mn    = fmaxf(mrow[r], t);
            const float alpha = exp2f((mrow[r] - mn) * LOG2E);
            mrow[r] = mn;
            const float e0 = exp2f((p0[r] - mn) * LOG2E);
            const float e1 = exp2f((p1[r] - mn) * LOG2E);
            p0[r] = e0; p1[r] = e1;
            float s = e0 + e1;
            s += __shfl_xor(s, 1, 32);
            s += __shfl_xor(s, 2, 32);
            s += __shfl_xor(s, 4, 32);
            s += __shfl_xor(s, 8, 32);
            lrow[r] = lrow[r] * alpha + s;
            o0[r] *= alpha; o1[r] *= alpha; o2[r] *= alpha; o3[r] *= alpha;
        }

        // ---- stage P (C layout -> A layout) through per-wave LDS ----
#pragma unroll
        for (int r = 0; r < 8; ++r) {
            const int rr = r + 8 * kh;
            pw[rr * 32 + ln16]      = f2bf(p0[r]);
            pw[rr * 32 + 16 + ln16] = f2bf(p1[r]);
        }
        Frag16 pa;   // lane row = ln16, K-half = kh  (wave-local LDS: DS ops are in-order per wave)
        pa.q[0] = *(const u32x4*)(pw + ln16 * 32 + kh * 16);
        pa.q[1] = *(const u32x4*)(pw + ln16 * 32 + kh * 16 + 8);

        // ---- P x V : 4 dh-blocks, K-dim = 32 keys ----
        {
            const unsigned short* vr = &Vt_lds[(0  + ln16) * SKV_PAD + n0 + kh * 16];
            Frag16 bv; bv.q[0] = *(const u32x4*)vr; bv.q[1] = *(const u32x4*)(vr + 8);
            o0 = __builtin_amdgcn_wmma_f32_16x16x32_bf16(false, pa.v, false, bv.v, (short)0, o0, false, false);
        }
        {
            const unsigned short* vr = &Vt_lds[(16 + ln16) * SKV_PAD + n0 + kh * 16];
            Frag16 bv; bv.q[0] = *(const u32x4*)vr; bv.q[1] = *(const u32x4*)(vr + 8);
            o1 = __builtin_amdgcn_wmma_f32_16x16x32_bf16(false, pa.v, false, bv.v, (short)0, o1, false, false);
        }
        {
            const unsigned short* vr = &Vt_lds[(32 + ln16) * SKV_PAD + n0 + kh * 16];
            Frag16 bv; bv.q[0] = *(const u32x4*)vr; bv.q[1] = *(const u32x4*)(vr + 8);
            o2 = __builtin_amdgcn_wmma_f32_16x16x32_bf16(false, pa.v, false, bv.v, (short)0, o2, false, false);
        }
        {
            const unsigned short* vr = &Vt_lds[(48 + ln16) * SKV_PAD + n0 + kh * 16];
            Frag16 bv; bv.q[0] = *(const u32x4*)vr; bv.q[1] = *(const u32x4*)(vr + 8);
            o3 = __builtin_amdgcn_wmma_f32_16x16x32_bf16(false, pa.v, false, bv.v, (short)0, o3, false, false);
        }
    }

    // ---------------- epilogue: O / l, write fp32 output (NT stores, write-once) ----------------
#pragma unroll
    for (int r = 0; r < 8; ++r) {
        const int qrow_c = q0 + w * 16 + r + 8 * kh;
        const float inv  = 1.0f / lrow[r];
        float* op = out + ((size_t)b * SQ + qrow_c) * Dd + h * DH + ln16;
        __builtin_nontemporal_store(o0[r] * inv, op);
        __builtin_nontemporal_store(o1[r] * inv, op + 16);
        __builtin_nontemporal_store(o2[r] * inv, op + 32);
        __builtin_nontemporal_store(o3[r] * inv, op + 48);
    }
}

extern "C" void kernel_launch(void* const* d_in, const int* in_sizes, int n_in,
                              void* d_out, int out_size, void* d_ws, size_t ws_size,
                              hipStream_t stream) {
    (void)in_sizes; (void)n_in; (void)out_size; (void)d_ws; (void)ws_size;
    const float*         q   = (const float*)d_in[0];
    const float*         k   = (const float*)d_in[1];
    const float*         v   = (const float*)d_in[2];
    const unsigned char* mb  = (const unsigned char*)d_in[3];   // jax bool -> 1 byte
    const float*         msc = (const float*)d_in[4];
    // d_in[5] = heads (scalar, == 10) — shapes are compile-time constants here.
    float* out = (float*)d_out;

    const size_t shmem = (size_t)LDS_US * sizeof(unsigned short);   // 128 KB dynamic LDS
    (void)hipFuncSetAttribute((const void*)omost_dd_attn_kernel,
                              hipFuncAttributeMaxDynamicSharedMemorySize, (int)shmem);

    dim3 grid(SQ / QTILE, H, Bb);   // (32, 10, 2)
    dim3 block(NTHR);               // 256 threads = 8 wave32
    omost_dd_attn_kernel<<<grid, block, shmem, stream>>>(q, k, v, mb, msc, out);
}
